// unit_gcn_901943132469
// MI455X (gfx1250) — matrix operand order repaired and checked by reading, not compiled, not run on hardware
//
#include <hip/hip_runtime.h>

// ---------------------------------------------------------------------------
// unit_gcn on MI455X (gfx1250):
//   out = ReLU(BN(merged graph-conv))   ==   one GEMM + fused BN stats
// GEMM: (19200 x 1600) = (19200 x 1600) x (1600 x 1600)
//   rows = (n,t), cols = (v*64+o), K = (u*64+c);  98.3 GFLOP, ~250 MB traffic
//   -> compute bound -> bf16 WMMA, f32 accumulate.
// Bias is constant per channel and cancels under BatchNorm -> dropped.
// v2: 32Mx64N wave tile (8 WMMA / k-step / wave), double-buffered LDS so
//     global_load_b128 of tile k+1 overlaps the WMMA chain of tile k.
// ---------------------------------------------------------------------------

typedef __bf16 v16bf __attribute__((ext_vector_type(16)));
typedef float  v8f   __attribute__((ext_vector_type(8)));

#define NB   64
#define CC   64
#define TT   300
#define VV   25
#define OO   64
#define KDIM 1600    // V*C
#define NCOL 1600    // V*O
#define NROW 19200   // N*T
#define CNT_PER_CH 480000.0f   // N*T*V

__device__ __forceinline__ unsigned short f2bf(float f) {
    unsigned u = __float_as_uint(f);
    unsigned r = (u + 0x7FFFu + ((u >> 16) & 1u)) >> 16;  // round-nearest-even
    return (unsigned short)r;
}

// -------- kernel 1: x(N,C,T,V) f32 -> xb[n][t][u][c] bf16; zero stats --------
__global__ __launch_bounds__(256) void k_xpose(const float* __restrict__ x,
                                               unsigned short* __restrict__ xb,
                                               float* __restrict__ gstats) {
    int j  = blockIdx.x * 256 + threadIdx.x;          // < 30,720,000
    int c  = j & 63;
    int r  = j >> 6;                                  // n*7500 + (t*25+u)
    int pu = r % 7500;
    int n  = r / 7500;
    xb[j] = f2bf(x[(((size_t)n << 6) + c) * 7500 + pu]);
    if (blockIdx.x == 0 && threadIdx.x < 128) gstats[threadIdx.x] = 0.0f;
}

// -------- kernel 2: merged weights, stored B^T: Mtt[(v*64+o)][(u*64+c)] -----
__global__ __launch_bounds__(256) void k_weights(const float* __restrict__ W,
                                                 unsigned short* __restrict__ Mtt) {
    int j = blockIdx.x * 256 + threadIdx.x;           // < 2,560,000
    int c = j & 63;
    int u = (j >> 6) % 25;
    int o = (j / 1600) & 63;
    int v = j / 102400;
    float val;
    if (u == v) {                                     // identity tap: sum_i W0_i
        val = 0.0f;
        #pragma unroll 4
        for (int i = 0; i < 24; ++i)
            val += W[((((i << 6) + o) << 6) + c) * 2 + 0];
    } else {                                          // circular tap
        int i = (u - v - 1 + 25) % 25;                // in [0,23] when u!=v
        val = W[((((i << 6) + o) << 6) + c) * 2 + 1];
    }
    Mtt[j] = f2bf(val);
}

// -------- kernel 3: bf16 WMMA GEMM + fused BN-stat accumulation -------------
// block tile 256(M) x 64(N); 8 waves, each owns 32Mx64N (2 A-frags x 4 B-frags
// = 8 WMMA per k-step). Double-buffered LDS, one barrier per k-step.
#define PITA 40   // 32 k + 8 bf16 pad: 80B pitch, all ds_*_b128 16B-aligned
#define PITB 40
#define ASZ  (256 * PITA)
#define BSZ  (64 * PITB)

__global__ __launch_bounds__(256) void k_gemm(const unsigned short* __restrict__ xb,
                                              const unsigned short* __restrict__ Mtt,
                                              float* __restrict__ out,
                                              float* __restrict__ gstats) {
    __shared__ __align__(16) unsigned short As[2][ASZ];
    __shared__ __align__(16) unsigned short Bs[2][BSZ];
    __shared__ float sstat[128];

    const int tid   = threadIdx.x;
    const int wave  = tid >> 5;
    const int lane  = tid & 31;
    const int laneM = lane & 15;
    const int hi    = lane >> 4;           // lane half -> K sub-range (ISA layout)
    const int rowBase = blockIdx.x * 256;  // 19200/256 = 75
    const int colBase = blockIdx.y * 64;   // 1600/64  = 25 (col block == vertex v)

    if (tid < 128) sstat[tid] = 0.0f;

    v8f acc[2][4];
    #pragma unroll
    for (int mi = 0; mi < 2; ++mi)
        #pragma unroll
        for (int j = 0; j < 4; ++j) acc[mi][j] = v8f{};

    // cooperative-load assignments
    const int brow = tid >> 2, bpart = tid & 3;             // B: 64 x 32 bf16
    const uint4* gA = reinterpret_cast<const uint4*>(
        xb + (size_t)(rowBase + tid) * KDIM);               // A: one 32k-row/thread
    const uint4* gB = reinterpret_cast<const uint4*>(
        Mtt + (size_t)(colBase + brow) * KDIM + bpart * 8);

    // prologue: fetch tile 0 into registers
    uint4 ra0 = gA[0], ra1 = gA[1], ra2 = gA[2], ra3 = gA[3];
    uint4 rb0 = gB[0];

    int buf = 0;
    for (int kk = 0; kk < KDIM; kk += 32) {
        // registers -> LDS[buf]
        unsigned short* as = &As[buf][tid * PITA];
        *reinterpret_cast<uint4*>(as + 0)  = ra0;
        *reinterpret_cast<uint4*>(as + 8)  = ra1;
        *reinterpret_cast<uint4*>(as + 16) = ra2;
        *reinterpret_cast<uint4*>(as + 24) = ra3;
        *reinterpret_cast<uint4*>(&Bs[buf][brow * PITB + bpart * 8]) = rb0;
        __syncthreads();

        // prefetch tile kk+32 (overlaps the WMMA chain below)
        if (kk + 32 < KDIM) {
            const uint4* pa = reinterpret_cast<const uint4*>(
                xb + (size_t)(rowBase + tid) * KDIM + (kk + 32));
            ra0 = pa[0]; ra1 = pa[1]; ra2 = pa[2]; ra3 = pa[3];
            rb0 = *reinterpret_cast<const uint4*>(
                Mtt + (size_t)(colBase + brow) * KDIM + (kk + 32) + bpart * 8);
        }

        // A fragments (16x32 bf16): lanes<16 K 0-7 & 16-23, lanes>=16 K 8-15 & 24-31
        union { uint4 u[2]; v16bf v; } fa[2];
        #pragma unroll
        for (int mi = 0; mi < 2; ++mi) {
            const unsigned short* ab =
                &As[buf][(wave * 32 + mi * 16 + laneM) * PITA + hi * 8];
            fa[mi].u[0] = *reinterpret_cast<const uint4*>(ab);
            fa[mi].u[1] = *reinterpret_cast<const uint4*>(ab + 16);
        }
        // B fragments (32x16 bf16): lanes<16 K 0-7 & 16-23, lanes>=16 K 8-15 & 24-31
        union { uint4 u[2]; v16bf v; } fb[4];
        #pragma unroll
        for (int j = 0; j < 4; ++j) {
            const unsigned short* bb = &Bs[buf][(j * 16 + laneM) * PITB + hi * 8];
            fb[j].u[0] = *reinterpret_cast<const uint4*>(bb);
            fb[j].u[1] = *reinterpret_cast<const uint4*>(bb + 16);
        }
        // 8 back-to-back WMMAs (A and B frags all resident)
        #pragma unroll
        for (int mi = 0; mi < 2; ++mi)
            #pragma unroll
            for (int j = 0; j < 4; ++j)
                acc[mi][j] = __builtin_amdgcn_wmma_f32_16x16x32_bf16(
                    false, fa[mi].v, false, fb[j].v, (short)0, acc[mi][j],
                    false, false);
        buf ^= 1;
    }

    // epilogue: scatter to (N,O,T,V) + per-channel sum / sumsq
    const int v = blockIdx.y;                 // all 64 cols of this block share v
    #pragma unroll
    for (int j = 0; j < 4; ++j) {
        const int o = j * 16 + laneM;         // colBase is 64-aligned
        float s1 = 0.0f, s2 = 0.0f;
        #pragma unroll
        for (int mi = 0; mi < 2; ++mi) {
            #pragma unroll
            for (int i = 0; i < 8; ++i) {     // C/D: lanes<16 M=i, lanes>=16 M=8+i
                int r = rowBase + wave * 32 + mi * 16 + hi * 8 + i;
                int n = r / TT, t = r - n * TT;
                float val = acc[mi][j][i];
                out[((((size_t)n << 6) + o) * TT + t) * VV + v] = val;
                s1 += val;
                s2 += val * val;
            }
        }
        atomicAdd(&sstat[o], s1);             // ds_add_f32
        atomicAdd(&sstat[64 + o], s2);
    }
    __syncthreads();
    if (tid < 128) atomicAdd(&gstats[tid], sstat[tid]);
}

// -------- kernel 4: BatchNorm (batch stats) + ReLU, float4-vectorized -------
__global__ __launch_bounds__(256) void k_bn(float* __restrict__ out,
                                            const float* __restrict__ gstats,
                                            const float* __restrict__ gamma,
                                            const float* __restrict__ beta) {
    int i = blockIdx.x * 256 + threadIdx.x;   // float4 index, 7,680,000 total
    int o = ((i * 4) / 7500) & 63;            // 7500 % 4 == 0 -> one channel / float4
    float m   = gstats[o] / CNT_PER_CH;
    float var = gstats[64 + o] / CNT_PER_CH - m * m;
    float inv = rsqrtf(var + 1e-5f);
    float sc  = gamma[o] * inv;
    float sh  = beta[o] - m * sc;
    float4* p = reinterpret_cast<float4*>(out) + i;
    float4 q = *p;
    q.x = fmaxf(fmaf(q.x, sc, sh), 0.0f);
    q.y = fmaxf(fmaf(q.y, sc, sh), 0.0f);
    q.z = fmaxf(fmaf(q.z, sc, sh), 0.0f);
    q.w = fmaxf(fmaf(q.w, sc, sh), 0.0f);
    *p = q;
}

extern "C" void kernel_launch(void* const* d_in, const int* in_sizes, int n_in,
                              void* d_out, int out_size, void* d_ws, size_t ws_size,
                              hipStream_t stream) {
    const float* x     = (const float*)d_in[0];
    const float* W     = (const float*)d_in[1];
    // d_in[2] (bias) intentionally unused: constant per channel -> cancels in BN
    const float* gamma = (const float*)d_in[3];
    const float* beta  = (const float*)d_in[4];
    float* out = (float*)d_out;

    unsigned short* xb  = (unsigned short*)d_ws;                 // 30,720,000 bf16
    unsigned short* Mtt = xb + (size_t)NROW * KDIM;              //  2,560,000 bf16
    float* gstats = (float*)((char*)d_ws + (size_t)NROW * KDIM * 2
                                         + (size_t)NCOL * KDIM * 2); // 128 f32

    k_xpose  <<<120000, 256, 0, stream>>>(x, xb, gstats);
    k_weights<<< 10000, 256, 0, stream>>>(W, Mtt);
    k_gemm   <<<dim3(NROW / 256, NCOL / 64), 256, 0, stream>>>(xb, Mtt, out, gstats);
    k_bn     <<< 30000, 256, 0, stream>>>(out, gstats, gamma, beta);
}
